// TemporalSelfAttention_72129680769616
// MI455X (gfx1250) — matrix operand order repaired and checked by reading, not compiled, not run on hardware
//
#include <hip/hip_runtime.h>
#include <cstdint>
#include <cstddef>

typedef __attribute__((ext_vector_type(8)))  __bf16 v8bf;
typedef __attribute__((ext_vector_type(16))) __bf16 v16bf;
typedef __attribute__((ext_vector_type(8)))  float  v8f;

namespace {
constexpr int kB  = 2;
constexpr int kS  = 2048;
constexpr int kD  = 1024;
constexpr int kH  = 16;
constexpr int kHd = 64;
constexpr float kInvScale = 0.125f;   // 1/sqrt(64)
}

__device__ __forceinline__ v16bf cat16(v8bf lo, v8bf hi) {
  return __builtin_shufflevector(lo, hi, 0,1,2,3,4,5,6,7,8,9,10,11,12,13,14,15);
}

__device__ __forceinline__ v8f wmma_bf16(v16bf a, v16bf b, v8f c) {
  return __builtin_amdgcn_wmma_f32_16x16x32_bf16(false, a, false, b, (short)0, c,
                                                 false, false);
}

// CDNA5 async global->LDS copy (ASYNCcnt-tracked, GVS addressing mode).
// lds_dst: generic pointer into __shared__; low 32 bits are the LDS byte addr.
__device__ __forceinline__ void async_copy_b128(void* lds_dst, const void* gbase,
                                                uint32_t byte_off) {
  uint32_t lds = (uint32_t)(uintptr_t)lds_dst;
  asm volatile("global_load_async_to_lds_b128 %0, %1, %2"
               :: "v"(lds), "v"(byte_off), "s"(gbase)
               : "memory");
}

#define WAIT_ASYNC_0() asm volatile("s_wait_asynccnt 0" ::: "memory")
#define WAIT_ASYNC_1() asm volatile("s_wait_asynccnt 1" ::: "memory")
#define WAIT_ASYNC_2() asm volatile("s_wait_asynccnt 2" ::: "memory")

// ---------------------------------------------------------------------------
// K0: fp32 -> bf16 cast
// ---------------------------------------------------------------------------
__global__ void cast_bf16_kernel(const float* __restrict__ src,
                                 __bf16* __restrict__ dst, int n) {
  int i = blockIdx.x * blockDim.x + threadIdx.x;
  if (i < n) dst[i] = (__bf16)src[i];
}

// ---------------------------------------------------------------------------
// K1: out[n,o] = sum_k A[n,k] * W[o,k] + bias[o], stored head-major bf16.
// 256 threads (8 waves). Block tile M=128, N=64, K-step 32, double-buffered
// LDS with async global->LDS for the A tile.
// ---------------------------------------------------------------------------
__global__ __launch_bounds__(256)
void qkv_gemm_kernel(const __bf16* __restrict__ A,     // (4096,1024) row-major
                     const __bf16* __restrict__ W,     // (1024,1024) row-major [o][k]
                     const float*  __restrict__ bias,  // (1024)
                     __bf16* __restrict__ outHM) {     // (B,H,S,Hd)
  __shared__ __bf16 At[2][128][32];
  __shared__ __bf16 Wt[2][32][64];

  const int tid   = threadIdx.x;
  const int lane  = tid & 31;
  const int wave  = tid >> 5;
  const int obase = blockIdx.x * 64;
  const int mbase = blockIdx.y * 128;

  const v8f vzero = {0.f,0.f,0.f,0.f,0.f,0.f,0.f,0.f};
  v8f acc[4] = {vzero, vzero, vzero, vzero};

  const int ar  = tid >> 1;          // 0..127
  const int ac  = (tid & 1) * 16;    // 0 or 16
  const int wo  = tid >> 2;          // 0..63
  const int wk8 = (tid & 3) * 8;     // 0,8,16,24

  auto stageA = [&](int buf, int k0) {
    uint32_t off = (uint32_t)(((size_t)(mbase + ar) * kD + k0 + ac) * 2u);
    async_copy_b128(&At[buf][ar][ac],     A, off);
    async_copy_b128(&At[buf][ar][ac + 8], A, off + 16u);
  };
  auto stageW = [&](int buf, int k0) {  // transpose while storing to LDS
    v8bf w8 = *(const v8bf*)&W[(size_t)(obase + wo) * kD + k0 + wk8];
    #pragma unroll
    for (int j = 0; j < 8; ++j) Wt[buf][wk8 + j][wo] = w8[j];
  };

  stageA(0, 0);
  stageW(0, 0);

  for (int k0 = 0; k0 < kD; k0 += 32) {
    const int cur = (k0 >> 5) & 1;
    __syncthreads();                        // all waves done reading buf cur^1
    if (k0 + 32 < kD) {
      stageA(cur ^ 1, k0 + 32);
      stageW(cur ^ 1, k0 + 32);
      if (k0 + 64 < kD)
        __builtin_prefetch(&W[(size_t)(obase + wo) * kD + k0 + 64 + wk8], 0, 3);
      WAIT_ASYNC_2();                       // drain batch for `cur`, keep next in flight
    } else {
      WAIT_ASYNC_0();
    }
    __syncthreads();                        // staged data visible to all waves

    // A fragment: lane l<16 -> row r, K {0..7,16..23}; l>=16 -> {8..15,24..31}
    const int r  = wave * 16 + (lane & 15);
    const int h8 = (lane >> 4) * 8;
    v16bf a = cat16(*(const v8bf*)&At[cur][r][h8],
                    *(const v8bf*)&At[cur][r][16 + h8]);
    v16bf bf[4];
    #pragma unroll
    for (int t = 0; t < 4; ++t)
      #pragma unroll
      for (int s = 0; s < 16; ++s)
        bf[t][s] = Wt[cur][(s < 8 ? h8 + s : 8 + h8 + s)][t * 16 + (lane & 15)];
    #pragma unroll
    for (int t = 0; t < 4; ++t) acc[t] = wmma_bf16(a, bf[t], acc[t]);
  }

  // Epilogue: C layout — elem j -> row j + 8*(lane/16); col = lane&15
  const int h = blockIdx.x;  // obase / Hd, since N tile == head width
  #pragma unroll
  for (int t = 0; t < 4; ++t) {
    #pragma unroll
    for (int j = 0; j < 8; ++j) {
      int row = mbase + wave * 16 + j + ((lane >> 4) << 3);
      int d   = t * 16 + (lane & 15);
      int bb  = row >> 11;        // / kS
      int ss  = row & (kS - 1);
      float v = acc[t][j] + bias[obase + d];
      outHM[(((size_t)bb * kH + h) * kS + ss) * kHd + d] = (__bf16)v;
    }
  }
}

// ---------------------------------------------------------------------------
// K2: flash attention. 256 threads per (b, h, 128 q rows). Each wave owns a
// 16-row q tile; K/V streamed in 32-row chunks, double-buffered, V via async.
// ---------------------------------------------------------------------------
__global__ __launch_bounds__(256)
void attention_kernel(const __bf16* __restrict__ Q,
                      const __bf16* __restrict__ Kh,
                      const __bf16* __restrict__ Vh,
                      const int*    __restrict__ mask,
                      __bf16* __restrict__ att) {
  __shared__ __bf16 Kt[2][64][32];     // K^T chunk: [d][k]  (B operand of Q@K^T)
  __shared__ __bf16 Vt[2][32][64];     // V chunk:   [k][d]  (B operand of P@V)
  __shared__ __bf16 Pb[8][16][40];     // per-wave P bounce (80B row stride)
  __shared__ int    mk[2][32];

  const int tid  = threadIdx.x;
  const int lane = tid & 31;
  const int wave = tid >> 5;
  const int qb   = blockIdx.x * 128;
  const int h    = blockIdx.y;
  const int b    = blockIdx.z;
  const size_t headoff = ((size_t)b * kH + h) * (size_t)kS * kHd;

  const int h8 = (lane >> 4) * 8;
  const int qr = qb + wave * 16 + (lane & 15);

  // Q fragments: two K=32 chunks over Hd=64
  v16bf aq[2];
  #pragma unroll
  for (int c = 0; c < 2; ++c) {
    v8bf lo = *(const v8bf*)&Q[headoff + (size_t)qr * kHd + c * 32 + h8];
    v8bf hi = *(const v8bf*)&Q[headoff + (size_t)qr * kHd + c * 32 + 16 + h8];
    aq[c] = cat16(lo, hi);
  }

  const v8f vzero = {0.f,0.f,0.f,0.f,0.f,0.f,0.f,0.f};
  v8f acc[4] = {vzero, vzero, vzero, vzero};
  float m[8], lsum[8];
  #pragma unroll
  for (int j = 0; j < 8; ++j) { m[j] = -3.0e38f; lsum[j] = 0.f; }

  const int skr = tid >> 3;        // 0..31 : k row in chunk
  const int sd8 = (tid & 7) * 8;   // 0..56 : d offset

  auto stageKV = [&](int buf, int k0) {
    // K: transpose while storing (scattered bf16 stores)
    v8bf kv = *(const v8bf*)&Kh[headoff + (size_t)(k0 + skr) * kHd + sd8];
    #pragma unroll
    for (int j = 0; j < 8; ++j) Kt[buf][sd8 + j][skr] = kv[j];
    // V: async DMA straight into LDS (row-major, no transpose needed)
    uint32_t voff = (uint32_t)((headoff + (size_t)(k0 + skr) * kHd + sd8) * 2u);
    async_copy_b128(&Vt[buf][skr][sd8], Vh, voff);
    if (tid < 32) mk[buf][tid] = mask[b * kS + k0 + tid];
  };

  stageKV(0, 0);

  for (int k0 = 0; k0 < kS; k0 += 32) {
    const int cur = (k0 >> 5) & 1;
    __syncthreads();                        // done reading buf cur^1
    if (k0 + 32 < kS) {
      stageKV(cur ^ 1, k0 + 32);
      WAIT_ASYNC_1();                       // drain batch for `cur`
    } else {
      WAIT_ASYNC_0();
    }
    __syncthreads();

    // scores: two 16-col tiles, contraction over d in two 32-chunks
    v16bf bk[2][2];
    #pragma unroll
    for (int kn = 0; kn < 2; ++kn)
      #pragma unroll
      for (int c = 0; c < 2; ++c)
        #pragma unroll
        for (int s = 0; s < 16; ++s)
          bk[kn][c][s] = Kt[cur][c * 32 + (s < 8 ? h8 + s : 8 + h8 + s)][kn * 16 + (lane & 15)];
    v8f sc[2] = {vzero, vzero};
    #pragma unroll
    for (int kn = 0; kn < 2; ++kn)
      #pragma unroll
      for (int c = 0; c < 2; ++c)
        sc[kn] = wmma_bf16(aq[c], bk[kn][c], sc[kn]);

    // mask + scale
    #pragma unroll
    for (int kn = 0; kn < 2; ++kn) {
      bool dead = (mk[cur][kn * 16 + (lane & 15)] == 0);
      #pragma unroll
      for (int j = 0; j < 8; ++j)
        sc[kn][j] = dead ? -3.0e38f : sc[kn][j] * kInvScale;
    }

    // online softmax: each q-row lives on 16 lanes of one half-wave
    #pragma unroll
    for (int j = 0; j < 8; ++j) {
      float rmax = fmaxf(sc[0][j], sc[1][j]);
      #pragma unroll
      for (int off = 8; off > 0; off >>= 1)
        rmax = fmaxf(rmax, __shfl_xor(rmax, off, 32));
      float nm = fmaxf(m[j], rmax);
      float p0 = __expf(sc[0][j] - nm);
      float p1 = __expf(sc[1][j] - nm);
      float rs = p0 + p1;
      #pragma unroll
      for (int off = 8; off > 0; off >>= 1)
        rs += __shfl_xor(rs, off, 32);
      float alpha = __expf(m[j] - nm);
      m[j] = nm;
      lsum[j] = lsum[j] * alpha + rs;
      #pragma unroll
      for (int t = 0; t < 4; ++t) acc[t][j] *= alpha;
      int prow = j + ((lane >> 4) << 3);
      Pb[wave][prow][lane & 15]        = (__bf16)p0;
      Pb[wave][prow][16 + (lane & 15)] = (__bf16)p1;
    }
    // wave-private LDS bounce: ensure P stores land before fragment reload
    asm volatile("s_wait_dscnt 0" ::: "memory");

    // P fragment (A operand, K=32) + P@V
    v16bf ap = cat16(*(const v8bf*)&Pb[wave][lane & 15][h8],
                     *(const v8bf*)&Pb[wave][lane & 15][16 + h8]);
    v16bf bv[4];
    #pragma unroll
    for (int t = 0; t < 4; ++t)
      #pragma unroll
      for (int s = 0; s < 16; ++s)
        bv[t][s] = Vt[cur][(s < 8 ? h8 + s : 8 + h8 + s)][t * 16 + (lane & 15)];
    #pragma unroll
    for (int t = 0; t < 4; ++t) acc[t] = wmma_bf16(ap, bv[t], acc[t]);
  }

  // normalize and store attended, layout (B,S,D) with D = H*Hd concatenated
  #pragma unroll
  for (int j = 0; j < 8; ++j) {
    float inv = 1.0f / lsum[j];
    int q = qb + wave * 16 + j + ((lane >> 4) << 3);
    #pragma unroll
    for (int t = 0; t < 4; ++t) {
      int d = t * 16 + (lane & 15);
      att[((size_t)b * kS + q) * kD + h * kHd + d] = (__bf16)(acc[t][j] * inv);
    }
  }
}

// ---------------------------------------------------------------------------
// K3a: y = att @ Wo^T + bo + x   (fp32 out). Same tiling as K1.
// ---------------------------------------------------------------------------
__global__ __launch_bounds__(256)
void out_proj_kernel(const __bf16* __restrict__ A,     // att bf16 (4096,1024)
                     const __bf16* __restrict__ W,     // Wo bf16  (1024,1024)
                     const float*  __restrict__ bias,
                     const float*  __restrict__ x,
                     float* __restrict__ y) {
  __shared__ __bf16 At[2][128][32];
  __shared__ __bf16 Wt[2][32][64];

  const int tid   = threadIdx.x;
  const int lane  = tid & 31;
  const int wave  = tid >> 5;
  const int obase = blockIdx.x * 64;
  const int mbase = blockIdx.y * 128;

  const v8f vzero = {0.f,0.f,0.f,0.f,0.f,0.f,0.f,0.f};
  v8f acc[4] = {vzero, vzero, vzero, vzero};

  const int ar  = tid >> 1;
  const int ac  = (tid & 1) * 16;
  const int wo  = tid >> 2;
  const int wk8 = (tid & 3) * 8;

  auto stageA = [&](int buf, int k0) {
    uint32_t off = (uint32_t)(((size_t)(mbase + ar) * kD + k0 + ac) * 2u);
    async_copy_b128(&At[buf][ar][ac],     A, off);
    async_copy_b128(&At[buf][ar][ac + 8], A, off + 16u);
  };
  auto stageW = [&](int buf, int k0) {
    v8bf w8 = *(const v8bf*)&W[(size_t)(obase + wo) * kD + k0 + wk8];
    #pragma unroll
    for (int j = 0; j < 8; ++j) Wt[buf][wk8 + j][wo] = w8[j];
  };

  stageA(0, 0);
  stageW(0, 0);

  for (int k0 = 0; k0 < kD; k0 += 32) {
    const int cur = (k0 >> 5) & 1;
    __syncthreads();
    if (k0 + 32 < kD) {
      stageA(cur ^ 1, k0 + 32);
      stageW(cur ^ 1, k0 + 32);
      if (k0 + 64 < kD)
        __builtin_prefetch(&W[(size_t)(obase + wo) * kD + k0 + 64 + wk8], 0, 3);
      WAIT_ASYNC_2();
    } else {
      WAIT_ASYNC_0();
    }
    __syncthreads();

    const int r  = wave * 16 + (lane & 15);
    const int h8 = (lane >> 4) * 8;
    v16bf a = cat16(*(const v8bf*)&At[cur][r][h8],
                    *(const v8bf*)&At[cur][r][16 + h8]);
    v16bf bf[4];
    #pragma unroll
    for (int t = 0; t < 4; ++t)
      #pragma unroll
      for (int s = 0; s < 16; ++s)
        bf[t][s] = Wt[cur][(s < 8 ? h8 + s : 8 + h8 + s)][t * 16 + (lane & 15)];
    #pragma unroll
    for (int t = 0; t < 4; ++t) acc[t] = wmma_bf16(a, bf[t], acc[t]);
  }

  #pragma unroll
  for (int t = 0; t < 4; ++t) {
    #pragma unroll
    for (int j = 0; j < 8; ++j) {
      int row = mbase + wave * 16 + j + ((lane >> 4) << 3);
      int col = obase + t * 16 + (lane & 15);
      y[(size_t)row * kD + col] =
          acc[t][j] + bias[col] + x[(size_t)row * kD + col];
    }
  }
}

// ---------------------------------------------------------------------------
// K3b: LayerNorm over last dim (1024), one block per row.
// ---------------------------------------------------------------------------
__global__ __launch_bounds__(256)
void layernorm_kernel(const float* __restrict__ y,
                      const float* __restrict__ gamma,
                      const float* __restrict__ beta,
                      float* __restrict__ out) {
  __shared__ float s1[256];
  __shared__ float s2[256];
  const int row = blockIdx.x;
  const int tid = threadIdx.x;

  const float4 t4 = ((const float4*)(y + (size_t)row * kD))[tid];
  s1[tid] = t4.x + t4.y + t4.z + t4.w;
  s2[tid] = t4.x * t4.x + t4.y * t4.y + t4.z * t4.z + t4.w * t4.w;
  __syncthreads();
  for (int s = 128; s > 0; s >>= 1) {
    if (tid < s) { s1[tid] += s1[tid + s]; s2[tid] += s2[tid + s]; }
    __syncthreads();
  }
  const float mu   = s1[0] * (1.0f / kD);
  const float var  = s2[0] * (1.0f / kD) - mu * mu;
  const float rstd = rsqrtf(var + 1e-5f);

  const float4 g4 = ((const float4*)gamma)[tid];
  const float4 b4 = ((const float4*)beta)[tid];
  float4 o;
  o.x = (t4.x - mu) * rstd * g4.x + b4.x;
  o.y = (t4.y - mu) * rstd * g4.y + b4.y;
  o.z = (t4.z - mu) * rstd * g4.z + b4.z;
  o.w = (t4.w - mu) * rstd * g4.w + b4.w;
  ((float4*)(out + (size_t)row * kD))[tid] = o;
}

// ---------------------------------------------------------------------------
extern "C" void kernel_launch(void* const* d_in, const int* in_sizes, int n_in,
                              void* d_out, int out_size, void* d_ws, size_t ws_size,
                              hipStream_t stream) {
  const float* x     = (const float*)d_in[0];
  const int*   mask  = (const int*)  d_in[1];
  const float* Wq    = (const float*)d_in[2];
  const float* bq    = (const float*)d_in[3];
  const float* Wk    = (const float*)d_in[4];
  const float* bk    = (const float*)d_in[5];
  const float* Wv    = (const float*)d_in[6];
  const float* bv    = (const float*)d_in[7];
  const float* Wo    = (const float*)d_in[8];
  const float* bo    = (const float*)d_in[9];
  const float* gamma = (const float*)d_in[10];
  const float* beta  = (const float*)d_in[11];
  float* out = (float*)d_out;

  char* ws = (char*)d_ws;
  const size_t MB = 1ull << 20;
  __bf16* xb  = (__bf16*)(ws +  0 * MB);   // 8 MB  (B*S*D bf16)
  __bf16* Qb  = (__bf16*)(ws +  8 * MB);   // 8 MB  (B,H,S,Hd)
  __bf16* Kb  = (__bf16*)(ws + 16 * MB);   // 8 MB
  __bf16* Vb  = (__bf16*)(ws + 24 * MB);   // 8 MB
  __bf16* Ab  = (__bf16*)(ws + 32 * MB);   // 8 MB  attended (B,S,D)
  __bf16* wqb = (__bf16*)(ws + 40 * MB);   // 2 MB each
  __bf16* wkb = (__bf16*)(ws + 42 * MB);
  __bf16* wvb = (__bf16*)(ws + 44 * MB);
  __bf16* wob = (__bf16*)(ws + 46 * MB);
  float*  y   = (float*) (ws +  0 * MB);   // 16 MB, aliases xb+Qb (dead by then)

  const int nX = kB * kS * kD;   // 4,194,304
  const int nW = kD * kD;        // 1,048,576

  cast_bf16_kernel<<<(nX + 255) / 256, 256, 0, stream>>>(x,  xb,  nX);
  cast_bf16_kernel<<<(nW + 255) / 256, 256, 0, stream>>>(Wq, wqb, nW);
  cast_bf16_kernel<<<(nW + 255) / 256, 256, 0, stream>>>(Wk, wkb, nW);
  cast_bf16_kernel<<<(nW + 255) / 256, 256, 0, stream>>>(Wv, wvb, nW);
  cast_bf16_kernel<<<(nW + 255) / 256, 256, 0, stream>>>(Wo, wob, nW);

  dim3 ggrid(kD / 64, (kB * kS) / 128);          // (16, 32)
  qkv_gemm_kernel<<<ggrid, 256, 0, stream>>>(xb, wqb, bq, Qb);
  qkv_gemm_kernel<<<ggrid, 256, 0, stream>>>(xb, wkb, bk, Kb);
  qkv_gemm_kernel<<<ggrid, 256, 0, stream>>>(xb, wvb, bv, Vb);

  dim3 agrid(kS / 128, kH, kB);                  // (16, 16, 2)
  attention_kernel<<<agrid, 256, 0, stream>>>(Qb, Kb, Vb, mask, Ab);

  out_proj_kernel<<<ggrid, 256, 0, stream>>>(Ab, wob, bo, x, y);

  layernorm_kernel<<<kB * kS, 256, 0, stream>>>(y, gamma, beta, out);
}
